// Attention_51187420234360
// MI455X (gfx1250) — compile-verified, matrix operand order checked by reading, not run on hardware
//
#include <hip/hip_runtime.h>

// Problem sizes (fixed by the reference)
#define P_N 16384
#define H_N 256
#define E_N 64
#define A_N 64

// Workspace layout (float offsets)
constexpr int OFF_TPART  = 0;                     // 64 WGs * 512
constexpr int OFF_SPART  = 32768;                 // 256 WGs * 512
constexpr int OFF_TSCALE = 163840;
constexpr int OFF_TBIAS  = 164096;
constexpr int OFF_SSCALE = 164352;
constexpr int OFF_SBIAS  = 164608;
constexpr int OFF_CVEC   = 164864;                // 256
constexpr int OFF_VVEC   = 165120;                // 256
constexpr int OFF_C0     = 165376;                // 1 (padded)
constexpr int OFF_WT     = 165392;                // 256x256 bf16 (32768 float slots), 16B aligned
constexpr int OFF_ATTN   = 198160;                // P*E floats
constexpr int OFF_MX     = 1246736;               // 64
constexpr int OFF_DEN    = 1246800;               // 64

typedef __attribute__((ext_vector_type(16))) __bf16 bf16x16;
typedef __attribute__((ext_vector_type(8)))  float  f32x8;

union FragBF { bf16x16 v; unsigned u[8]; };

__device__ __forceinline__ unsigned short f2bf(float f) {
  unsigned u = __float_as_uint(f);
  unsigned r = u + 0x7FFFu + ((u >> 16) & 1u);   // round-to-nearest-even
  return (unsigned short)(r >> 16);
}

// ---------- Pass 1: deterministic partial stats (reads spat once) ----------
__global__ void k_stats(const float* __restrict__ temp,
                        const float* __restrict__ spat,
                        float* __restrict__ ws) {
  const int b = blockIdx.x, t = threadIdx.x;
  if (b < 64) {                       // temp_hidden [P,H]: column sums, 256 rows/WG
    float s1 = 0.f, s2 = 0.f;
    const int r0 = b * 256;
    for (int r = 0; r < 256; ++r) {
      float x = temp[(size_t)(r0 + r) * H_N + t];
      s1 += x; s2 += x * x;
    }
    ws[OFF_TPART + b * 512 + t]       = s1;
    ws[OFF_TPART + b * 512 + 256 + t] = s2;
  } else {                            // spat [P,H,E]: thread t owns h=t, 64 persons/WG
    const int j = b - 64;
    float s1 = 0.f, s2 = 0.f;
    const int p0 = j * 64;
    for (int p = 0; p < 64; ++p) {
      const float4* q = (const float4*)(spat + (size_t)(p0 + p) * 16384 + t * 64);
#pragma unroll
      for (int i = 0; i < 16; ++i) {
        float4 x = q[i];
        s1 += x.x + x.y + x.z + x.w;
        s2 += x.x * x.x + x.y * x.y + x.z * x.z + x.w * x.w;
      }
    }
    ws[OFF_SPART + j * 512 + t]       = s1;
    ws[OFF_SPART + j * 512 + 256 + t] = s2;
  }
}

// ---------- Tiny: W = w_temp^T @ w_spat (stored transposed, bf16), c, v, c0 ----------
__global__ void k_wprep(const float* __restrict__ w_temp, const float* __restrict__ b_temp,
                        const float* __restrict__ w_spat, const float* __restrict__ b_spat,
                        float* __restrict__ ws) {
  const int b = blockIdx.x, t = threadIdx.x;
  if (b < 256) {
    float acc = 0.f;
    for (int a = 0; a < A_N; ++a) acc = fmaf(w_temp[a * H_N + t], w_spat[a * H_N + b], acc);
    unsigned short* WT = (unsigned short*)(ws + OFF_WT);
    WT[b * 256 + t] = f2bf(acc);                 // WT[h2][h'] = W[h',h2]
    if (t == 0) {
      float cv = 0.f;
      for (int a = 0; a < A_N; ++a) cv = fmaf(b_temp[a], w_spat[a * H_N + b], cv);
      ws[OFF_CVEC + b] = cv;
    }
  } else {
    float vv = 0.f;
    for (int a = 0; a < A_N; ++a) vv = fmaf(b_spat[a], w_temp[a * H_N + t], vv);
    ws[OFF_VVEC + t] = vv;
    if (t == 0) {
      float c0 = 0.f;
      for (int a = 0; a < A_N; ++a) c0 = fmaf(b_spat[a], b_temp[a], c0);
      ws[OFF_C0] = c0;
    }
  }
}

// ---------- Fold stats into per-h affine scale/bias ----------
__global__ void k_finalize(const float* __restrict__ gamma, const float* __restrict__ beta,
                           float* __restrict__ ws) {
  const int h = threadIdx.x;
  float s1 = 0.f, s2 = 0.f;
  for (int i = 0; i < 64; ++i) { s1 += ws[OFF_TPART + i * 512 + h]; s2 += ws[OFF_TPART + i * 512 + 256 + h]; }
  float mean = s1 / (float)P_N;
  float var  = s2 / (float)P_N - mean * mean;
  float inv  = rsqrtf(var + 1e-5f) * gamma[h];
  ws[OFF_TSCALE + h] = inv;
  ws[OFF_TBIAS + h]  = beta[h] - mean * inv;
  s1 = 0.f; s2 = 0.f;
  for (int i = 0; i < 256; ++i) { s1 += ws[OFF_SPART + i * 512 + h]; s2 += ws[OFF_SPART + i * 512 + 256 + h]; }
  const float n = (float)P_N * (float)E_N;
  mean = s1 / n; var = s2 / n - mean * mean;
  inv  = rsqrtf(var + 1e-5f) * gamma[h];
  ws[OFF_SSCALE + h] = inv;
  ws[OFF_SBIAS + h]  = beta[h] - mean * inv;
}

// ---------- Pass 2 (fused): WMMA u = tn@W in LDS, then stream spat -> attn logits ----------
__global__ void __launch_bounds__(256)
k_attn(const float* __restrict__ temp, const float* __restrict__ spat,
       float* __restrict__ ws) {
  __shared__ __align__(16) unsigned short sWT[256 * 256];   // 128 KiB bf16, [h2][h']
  __shared__ __align__(16) unsigned short sA[64 * 256];     //  32 KiB bf16 tn rows
  __shared__ __align__(16) float          sU[64 * 256];     //  64 KiB f32 u rows
  __shared__ float sScl[256], sBia[256], sS[64];

  const int tid = threadIdx.x;
  const int p0  = blockIdx.x * 64;

  { // stage W
    const uint4* g = (const uint4*)(ws + OFF_WT);
    uint4* s = (uint4*)sWT;
    for (int i = tid; i < 8192; i += 256) s[i] = g[i];
  }
  { // stage normalized tn as bf16 (A operand)
    const float* tsc = ws + OFF_TSCALE;
    const float* tbi = ws + OFF_TBIAS;
    for (int idx = tid; idx < 64 * 256; idx += 256) {
      const int lp = idx >> 8, h = idx & 255;
      sA[idx] = f2bf(fmaf(temp[(size_t)(p0 + lp) * H_N + h], tsc[h], tbi[h]));
    }
  }
  sScl[tid] = ws[OFF_SSCALE + tid];
  sBia[tid] = ws[OFF_SBIAS + tid];
  __syncthreads();

  { // WMMA: 64x256 = (64x256 bf16) x (256x256 bf16), K-loop of 8, 8 waves x 8 tiles
    const int wave = tid >> 5, lane = tid & 31;
    const int lr = lane & 15, hi = lane >> 4;
    const int m = wave >> 1;                 // M tile 0..3
    const int nbase = (wave & 1) * 8;        // N tiles nbase..nbase+7
    const unsigned* A32 = (const unsigned*)sA;
    const unsigned* B32 = (const unsigned*)sWT;
    f32x8 acc[8];
#pragma unroll
    for (int nt = 0; nt < 8; ++nt) acc[nt] = (f32x8){0.f,0.f,0.f,0.f,0.f,0.f,0.f,0.f};
    for (int k0 = 0; k0 < 256; k0 += 32) {
      FragBF a;
      const int ak = k0 + hi * 8;
      const int ab = ((m * 16 + lr) * 256 + ak) >> 1;
#pragma unroll
      for (int i = 0; i < 4; ++i) { a.u[i] = A32[ab + i]; a.u[4 + i] = A32[ab + 8 + i]; }
#pragma unroll
      for (int nt = 0; nt < 8; ++nt) {
        FragBF bf;
        const int bb = (((nbase + nt) * 16 + lr) * 256 + ak) >> 1;
#pragma unroll
        for (int i = 0; i < 4; ++i) { bf.u[i] = B32[bb + i]; bf.u[4 + i] = B32[bb + 8 + i]; }
        acc[nt] = __builtin_amdgcn_wmma_f32_16x16x32_bf16(
            false, a.v, false, bf.v, (short)0, acc[nt], false, false);
      }
    }
    const float* cvec = ws + OFF_CVEC;
#pragma unroll
    for (int nt = 0; nt < 8; ++nt) {
      const int col = (nbase + nt) * 16 + lr;
      const float cv = cvec[col];
      const int rbase = m * 16 + hi * 8;     // C/D layout: VGPR i -> M=i (lo lanes), M=i+8 (hi lanes)
#pragma unroll
      for (int i = 0; i < 8; ++i) sU[(rbase + i) * 256 + col] = acc[nt][i] + cv;
    }
  }

  if (tid < 64) { // s[p] = tn . v + c0 (from global temp row; conflict-free)
    const float* tsc = ws + OFF_TSCALE;
    const float* tbi = ws + OFF_TBIAS;
    const float* vv  = ws + OFF_VVEC;
    const float* row = temp + (size_t)(p0 + tid) * H_N;
    float d = ws[OFF_C0];
    for (int h = 0; h < 256; ++h) d = fmaf(fmaf(row[h], tsc[h], tbi[h]), vv[h], d);
    sS[tid] = d;
  }
  __syncthreads();

  // stream spat slice: attn[p,e] = 8*(sum_h sr*u + s[p]); one (lp,e2) row per wave-iter
  float* attn = ws + OFF_ATTN;
  const int wave = tid >> 5, lane = tid & 31;
  for (int r = wave; r < 64 * 64; r += 8) {
    const int lp = r >> 6, e2 = r & 63;
    const float* base = spat + (size_t)(p0 + lp) * 16384 + e2 * 256;
    const float4 x0 = *(const float4*)(base + lane * 4);
    const float4 x1 = *(const float4*)(base + 128 + lane * 4);
    const int hA = e2 * 4 + (lane >> 4);     // h index of reshaped elements
    const int hB = hA + 2;
    const float sa = sScl[hA], ba = sBia[hA];
    const float sb = sScl[hB], bb = sBia[hB];
    const float* up = sU + lp * 256;
    const float4 u0 = *(const float4*)(up + lane * 4);
    const float4 u1 = *(const float4*)(up + 128 + lane * 4);
    float acc;
    acc = fmaf(fmaf(x0.x, sa, ba), u0.x, 0.f);
    acc = fmaf(fmaf(x0.y, sa, ba), u0.y, acc);
    acc = fmaf(fmaf(x0.z, sa, ba), u0.z, acc);
    acc = fmaf(fmaf(x0.w, sa, ba), u0.w, acc);
    acc = fmaf(fmaf(x1.x, sb, bb), u1.x, acc);
    acc = fmaf(fmaf(x1.y, sb, bb), u1.y, acc);
    acc = fmaf(fmaf(x1.z, sb, bb), u1.z, acc);
    acc = fmaf(fmaf(x1.w, sb, bb), u1.w, acc);
#pragma unroll
    for (int off = 16; off >= 1; off >>= 1) acc += __shfl_xor(acc, off, 32);
    if (lane == 0) attn[(size_t)(p0 + lp) * 64 + e2] = 8.0f * (acc + sS[lp]);
  }
}

// ---------- Softmax over persons (per column e2): max + sum(exp) ----------
__global__ void k_soft(float* __restrict__ ws) {
  __shared__ float red[256];
  const int e2 = blockIdx.x, t = threadIdx.x;
  const float* attn = ws + OFF_ATTN;
  float m = -3.0e38f;
  for (int p = t; p < P_N; p += 256) m = fmaxf(m, attn[(size_t)p * 64 + e2]);
  red[t] = m; __syncthreads();
  for (int s = 128; s > 0; s >>= 1) { if (t < s) red[t] = fmaxf(red[t], red[t + s]); __syncthreads(); }
  const float mx = red[0]; __syncthreads();
  float sum = 0.f;
  for (int p = t; p < P_N; p += 256) sum += expf(attn[(size_t)p * 64 + e2] - mx);
  red[t] = sum; __syncthreads();
  for (int s = 128; s > 0; s >>= 1) { if (t < s) red[t] += red[t + s]; __syncthreads(); }
  if (t == 0) { ws[OFF_MX + e2] = mx; ws[OFF_DEN + e2] = red[0]; }
}

// ---------- Pass 3: out[p,h] = sum_e sr[p,e,h]*wv[p,e] (third spat stream) ----------
__global__ void k_out(const float* __restrict__ spat, const float* __restrict__ ws,
                      float* __restrict__ out) {
  __shared__ float wv[64];
  __shared__ float sScl[256], sBia[256];
  const int p = blockIdx.x, t = threadIdx.x;
  sScl[t] = ws[OFF_SSCALE + t];
  sBia[t] = ws[OFF_SBIAS + t];
  if (t < 64)
    wv[t] = expf(ws[OFF_ATTN + (size_t)p * 64 + t] - ws[OFF_MX + t]) / ws[OFF_DEN + t];
  __syncthreads();
  const float* base = spat + (size_t)p * 16384;
  float acc = 0.f;
#pragma unroll 4
  for (int e2 = 0; e2 < 64; ++e2) {
    const int f = e2 * 256 + t;
    const int h = (e2 << 2) + (t >> 6);
    acc = fmaf(fmaf(base[f], sScl[h], sBia[h]), wv[e2], acc);
  }
  out[(size_t)p * 256 + t] = acc;
}

extern "C" void kernel_launch(void* const* d_in, const int* in_sizes, int n_in,
                              void* d_out, int out_size, void* d_ws, size_t ws_size,
                              hipStream_t stream) {
  const float* temp   = (const float*)d_in[0];
  const float* spat   = (const float*)d_in[1];
  const float* gamma  = (const float*)d_in[2];
  const float* beta   = (const float*)d_in[3];
  const float* w_temp = (const float*)d_in[4];
  const float* b_temp = (const float*)d_in[5];
  const float* w_spat = (const float*)d_in[6];
  const float* b_spat = (const float*)d_in[7];
  float* ws  = (float*)d_ws;
  float* out = (float*)d_out;

  k_stats   <<<320, 256, 0, stream>>>(temp, spat, ws);
  k_wprep   <<<257, 256, 0, stream>>>(w_temp, b_temp, w_spat, b_spat, ws);
  k_finalize<<<1,   256, 0, stream>>>(gamma, beta, ws);
  k_attn    <<<P_N / 64, 256, 0, stream>>>(temp, spat, ws);
  k_soft    <<<E_N, 256, 0, stream>>>(ws);
  k_out     <<<P_N, 256, 0, stream>>>(spat, ws, out);
}
